// P_Net_16647293239592
// MI455X (gfx1250) — matrix operand order repaired and checked by reading, not compiled, not run on hardware
//
#include <hip/hip_runtime.h>
#include <math.h>

typedef float v8f  __attribute__((ext_vector_type(8)));
typedef unsigned int v8u __attribute__((ext_vector_type(8)));
typedef __bf16 v16bf __attribute__((ext_vector_type(16)));

#define B_   32
#define N_   64
#define K_   64
#define D_   128
#define JC   16           // plane columns per block
#define NJ   8            // column chunks
#define KT   16           // k per block
#define NKT  4
#define EPS_ 1e-7f
#define ZSTR 136          // bf16 elems per (k,j) row: 128 + 8 pad -> 68 dwords, 68%64=4 banks
#define WSTR 68           // dwords per W row: 64 + 4 pad

static __device__ inline v16bf as_bf16(v8u x) {
    union { v8u u; v16bf b; } c; c.u = x; return c.b;
}

// Split floats a,b into packed-bf16 hi and lo dwords ([15:0]=a, [31:16]=b).
// Truncation split: hi is exact top bits, lo = fl(x - hi) -> total rel err ~2^-16.
static __device__ inline void split2(float a, float b, unsigned int& hpk, unsigned int& lpk) {
    const unsigned int ba = __float_as_uint(a), bb = __float_as_uint(b);
    const float ra = __uint_as_float(ba & 0xFFFF0000u);
    const float rb = __uint_as_float(bb & 0xFFFF0000u);
    hpk = (ba >> 16) | (bb & 0xFFFF0000u);
    lpk = (__float_as_uint(a - ra) >> 16) | (__float_as_uint(b - rb) & 0xFFFF0000u);
}

#define WMMA_BF16(A, Bv, C) \
    __builtin_amdgcn_wmma_f32_16x16x32_bf16(false, as_bf16(A), false, as_bf16(Bv), (short)0, (C), false, false)

// LDS: Z split [16k][16j][136i] bf16 hi+lo (2x68KB) + W split packed pairs (2x34KB) = 204KB
__global__ __launch_bounds__(256) void pnet_main(const float* __restrict__ x,
                                                 const float* __restrict__ R,
                                                 const float* __restrict__ W,
                                                 float* __restrict__ partial) {
    __shared__ unsigned short lds_zh[KT * JC * ZSTR];
    __shared__ unsigned short lds_zl[KT * JC * ZSTR];
    __shared__ unsigned int   lds_wh[D_ * WSTR];
    __shared__ unsigned int   lds_wl[D_ * WSTR];

    const int jc   = blockIdx.x;   // 0..7
    const int kt   = blockIdx.y;   // 0..3
    const int b    = blockIdx.z;   // 0..31
    const int tid  = threadIdx.x;
    const int wave = tid >> 5;     // 0..7
    const int lane = tid & 31;
    const int lrow = lane & 15;
    const int lhi  = lane >> 4;

    // ---- stage W split into LDS as packed bf16 column pairs ----
    for (int p = tid; p < D_ * (D_ / 2); p += 256) {
        const int i = p >> 6, c = p & 63;
        unsigned int h, l;
        split2(W[i * D_ + 2 * c], W[i * D_ + 2 * c + 1], h, l);
        lds_wh[i * WSTR + c] = h;
        lds_wl[i * WSTR + c] = l;
    }

    // ---- Phase 1: Z[k,i,j] = sum_n R[n,kg] * x[b,n,i,jg]  via 3x bf16 WMMA ----
    // A (R^T) 16x32 bf16: lane row = lane%16; vgpr v -> K pair K0,K0+1,
    // K0 = 16*(v>>2) + 2*(v&3) + 8*lhi   (per ISA 16-bit A layout)
    v8u a_rh[2], a_rl[2];
    {
        const int kg = kt * KT + lrow;
        #pragma unroll
        for (int s = 0; s < 2; ++s)
            #pragma unroll
            for (int v = 0; v < 8; ++v) {
                const int n0 = 32 * s + 16 * (v >> 2) + 2 * (v & 3) + 8 * lhi;
                unsigned int h, l;
                split2(R[n0 * K_ + kg], R[(n0 + 1) * K_ + kg], h, l);
                a_rh[s][v] = h; a_rl[s][v] = l;
            }
    }

    const float* xb = x + (size_t)b * N_ * D_ * D_;
    #pragma unroll 1
    for (int tt = 0; tt < 16; ++tt) {              // uniform trip count (no exec-mask loop)
        const int t = wave + 8 * tt;               // i row of the d x d plane
        const float* xcol = xb + (size_t)t * D_ + jc * JC + lrow;
        v8f c = {0.f, 0.f, 0.f, 0.f, 0.f, 0.f, 0.f, 0.f};
        #pragma unroll
        for (int s = 0; s < 2; ++s) {
            // B (x) 32x16 bf16: vgpr v -> rows 16*(v>>2)+2*(v&3)+8*lhi, +1 ; col = lane%16
            v8u bh, bl;
            #pragma unroll
            for (int v = 0; v < 8; ++v) {
                const int n0 = 32 * s + 16 * (v >> 2) + 2 * (v & 3) + 8 * lhi;
                unsigned int h, l;
                split2(xcol[(size_t)n0 * (D_ * D_)],
                       xcol[(size_t)(n0 + 1) * (D_ * D_)], h, l);
                bh[v] = h; bl[v] = l;
            }
            c = WMMA_BF16(a_rh[s], bh, c);
            c = WMMA_BF16(a_rh[s], bl, c);
            c = WMMA_BF16(a_rl[s], bh, c);
        }
        // store C split into Z, layout [k][j][i] (16-bit elems, padded rows)
        #pragma unroll
        for (int r = 0; r < 8; ++r) {
            const int idx = ((r + 8 * lhi) * JC + lrow) * ZSTR + t;
            const unsigned int cb = __float_as_uint(c[r]);
            const float hf = __uint_as_float(cb & 0xFFFF0000u);
            lds_zh[idx] = (unsigned short)(cb >> 16);
            lds_zl[idx] = (unsigned short)(__float_as_uint(c[r] - hf) >> 16);
        }
    }
    __syncthreads();

    // ---- Phase 2: per k: Y = relu(W @ Z_k); acc += (Y+EPS)^2  via 3x bf16 WMMA ----
    const unsigned int* zh32 = (const unsigned int*)lds_zh;
    const unsigned int* zl32 = (const unsigned int*)lds_zl;
    float acc[2] = {0.f, 0.f};
    #pragma unroll 1
    for (int it = 0; it < 8; ++it) {
        // A = W rows it*16..+15: preload packed hi/lo for 4 K-steps (reused for 2 k's)
        v8u awh[4], awl[4];
        const int i2 = it * 16 + lrow;
        #pragma unroll
        for (int s = 0; s < 4; ++s)
            #pragma unroll
            for (int v = 0; v < 8; ++v) {
                const int p = 16 * s + 8 * (v >> 2) + (v & 3) + 4 * lhi;  // K0/2
                awh[s][v] = lds_wh[i2 * WSTR + p];
                awl[s][v] = lds_wl[i2 * WSTR + p];
            }
        #pragma unroll
        for (int q = 0; q < 2; ++q) {
            const int zrow = ((2 * wave + q) * JC + lrow) * (ZSTR / 2);  // dwords
            v8f c = {0.f, 0.f, 0.f, 0.f, 0.f, 0.f, 0.f, 0.f};
            #pragma unroll
            for (int s = 0; s < 4; ++s) {
                // B (Z_k) 32x16: vgpr v -> i-pair index p = 16s + 8*(v>>2) + (v&3) + 4*lhi
                v8u bh, bl;
                #pragma unroll
                for (int v = 0; v < 8; ++v) {
                    const int p = 16 * s + 8 * (v >> 2) + (v & 3) + 4 * lhi;
                    bh[v] = zh32[zrow + p];
                    bl[v] = zl32[zrow + p];
                }
                c = WMMA_BF16(awh[s], bh, c);
                c = WMMA_BF16(awh[s], bl, c);
                c = WMMA_BF16(awl[s], bh, c);
            }
            #pragma unroll
            for (int r = 0; r < 8; ++r) {
                const float tv = fmaxf(c[r], 0.f) + EPS_;
                acc[q] += tv * tv;
            }
        }
    }

    // ---- deterministic reduction: each wave owns its 2 k values ----
    #pragma unroll
    for (int q = 0; q < 2; ++q) {
        float v = acc[q];
        #pragma unroll
        for (int off = 16; off > 0; off >>= 1) v += __shfl_xor(v, off, 32);
        if (lane == 0) {
            const int kg = kt * KT + 2 * wave + q;
            partial[(b * K_ + kg) * NJ + jc] = v;
        }
    }
}

__global__ void pnet_finish(const float* __restrict__ partial, float* __restrict__ out) {
    const int idx = blockIdx.x * blockDim.x + threadIdx.x;
    if (idx < B_ * K_) {
        float s = 0.f;
        #pragma unroll
        for (int c = 0; c < NJ; ++c) s += partial[idx * NJ + c];
        out[idx] = sqrtf(s);
    }
}

extern "C" void kernel_launch(void* const* d_in, const int* in_sizes, int n_in,
                              void* d_out, int out_size, void* d_ws, size_t ws_size,
                              hipStream_t stream) {
    const float* x = (const float*)d_in[0];   // [B, N, D, D]
    const float* R = (const float*)d_in[1];   // [N, K]
    const float* W = (const float*)d_in[2];   // [D, D]
    float* partial = (float*)d_ws;            // B*K*NJ floats = 64 KB scratch

    dim3 grid(NJ, NKT, B_);                   // 1024 blocks, 8 wave32 each
    pnet_main<<<grid, 256, 0, stream>>>(x, R, W, partial);
    pnet_finish<<<(B_ * K_ + 255) / 256, 256, 0, stream>>>(partial, (float*)d_out);
}
